// CoAttention_37838661878290
// MI455X (gfx1250) — matrix operand order repaired and checked by reading, not compiled, not run on hardware
//
#include <hip/hip_runtime.h>
#include <hip/hip_bf16.h>

// ---------------------------------------------------------------------------
// Hyperbolic co-attention for MI455X (gfx1250).
// Big batched GEMMs run on v_wmma_f32_16x16x32_f16 (f16 operands, f32 accum),
// with tiles staged into LDS by the Tensor Data Mover (tensor_load_to_lds,
// TENSORcnt-tracked, hardware LDS padding = bank-conflict padding).
// All radial Poincare-ball ops are fused per-row reduction kernels (wave32).
// ---------------------------------------------------------------------------

typedef __attribute__((ext_vector_type(16))) _Float16 v16h;
typedef __attribute__((ext_vector_type(8)))  float    v8f;

typedef unsigned int tdm_v4u __attribute__((ext_vector_type(4)));
typedef int          tdm_v8i __attribute__((ext_vector_type(8)));
typedef int          tdm_v4i __attribute__((ext_vector_type(4)));

#define DEVI __device__ __forceinline__

constexpr int BB = 128;   // batch
constexpr int SS = 512;   // sentence length (== T)
constexpr int DD = 100;   // feature dim
constexpr int DP = 128;   // padded feature dim (multiple of 32 for WMMA K)
constexpr int KK = 128;   // attention dim k

constexpr float MINV = 1e-15f;
constexpr float MAXN = 1.0f - 4e-3f;   // (1-EPS)/sqrt(c), c = 1

DEVI float artanh_(float x) {
  x = fminf(fmaxf(x, -1.f + 1e-7f), 1.f - 1e-7f);
  return 0.5f * (log1pf(x) - log1pf(-x));
}

DEVI float block_sum(float v, float* sbuf, int tid, int nthr) {
#pragma unroll
  for (int o = 16; o > 0; o >>= 1) v += __shfl_xor(v, o, 32);
  __syncthreads();
  if ((tid & 31) == 0) sbuf[tid >> 5] = v;
  __syncthreads();
  float t = 0.f;
  const int nw = nthr >> 5;
  for (int i = 0; i < nw; ++i) t += sbuf[i];
  return t;
}

DEVI float block_max(float v, float* sbuf, int tid, int nthr) {
#pragma unroll
  for (int o = 16; o > 0; o >>= 1) v = fmaxf(v, __shfl_xor(v, o, 32));
  __syncthreads();
  if ((tid & 31) == 0) sbuf[tid >> 5] = v;
  __syncthreads();
  float t = -3.4e38f;
  const int nw = nthr >> 5;
  for (int i = 0; i < nw; ++i) t = fmaxf(t, sbuf[i]);
  return t;
}

// ------------------------- Tensor Data Mover helper -------------------------
#if defined(__gfx1250__) && __has_builtin(__builtin_amdgcn_tensor_load_to_lds)
#define USE_TDM 1
// 2D f16 tile load: rows x cols elements, global row stride = strideH elements,
// deposited contiguously into LDS with hardware padding (padInt/padAmt codes:
// pad_interval 2^(code+1) dwords, pad_amount (code+1) dwords).
DEVI void tdm_load_2d_f16(const _Float16* g, unsigned ldsByte, int rows, int cols,
                          int strideH, int padInt, int padAmt) {
  const unsigned long long ga = (unsigned long long)g;
  tdm_v4u g0;
  g0[0] = 1u;                                                    // count=1
  g0[1] = ldsByte;                                               // lds_addr
  g0[2] = (unsigned)ga;                                          // global_addr lo
  g0[3] = (unsigned)((ga >> 32) & 0x01FFFFFFull) | (2u << 30);   // addr[56:32]|type=2
  tdm_v8i g1;
  g1[0] = (int)((1u << 16)                 // data_size = 2 bytes
                | (1u << 20)               // pad_enable
                | ((unsigned)padInt << 22) // pad_interval code
                | ((unsigned)padAmt << 25));  // pad_amount code
  g1[1] = (int)(((unsigned)cols & 0xFFFFu) << 16);  // tensor_dim0 (bits 79:48 lo)
  g1[2] = (int)(((unsigned)rows & 0xFFFFu) << 16);  // tensor_dim1 (bits 111:80 lo)
  g1[3] = (int)(((unsigned)cols) << 16);            // tile_dim0 (bits 127:112)
  g1[4] = rows;                                     // tile_dim1 (tile_dim2 = 0)
  g1[5] = strideH;                                  // tensor_dim0_stride lo
  g1[6] = 0;
  g1[7] = 0;
  tdm_v4i z4 = {0, 0, 0, 0};
#if defined(__clang_major__) && (__clang_major__ >= 23)
  tdm_v8i z8 = {0, 0, 0, 0, 0, 0, 0, 0};
  __builtin_amdgcn_tensor_load_to_lds(g0, g1, z4, z4, z8, 0);
#else
  __builtin_amdgcn_tensor_load_to_lds(g0, g1, z4, z4, 0);
#endif
}
#else
#define USE_TDM 0
#endif

// ---------------------------------------------------------------------------
// Batched GEMM: C[b](MxN, f32) = A[b](MxKd, f16 row-major) * B[b](KdxN, f16)
// TRANSB=1: B stored as [N x Kd] row-major (i.e. logical B transposed).
// Block tile 128x128x32, 8 waves (2 M x 4 N), wave tile 64x32 (8 WMMA accums).
// LDS: A [128][36] row-major; B: TRANS -> [128][36] n-major (pair loads),
//      non-TRANS -> [32][132] k-major (strided 16-bit frag loads).
// ---------------------------------------------------------------------------
union HFrag { v16h v; _Float16 h[16]; unsigned u[8]; };
union CFrag { v8f v; float f[8]; };

template <int TRANSB>
__global__ __launch_bounds__(256) void k_gemm_f16(
    const _Float16* __restrict__ A, long long sA, int lda,
    const _Float16* __restrict__ Bm, long long sB, int ldb,
    float* __restrict__ C, long long sC, int N, int Kd) {
  constexpr int LDT = 36;   // A (and trans-B) LDS leading dim in halves
  constexpr int LDBN = 132; // non-trans B LDS leading dim (128 + 4 pad)
  __shared__ __align__(16) _Float16 Ash[128 * LDT];
  __shared__ __align__(16) _Float16 Bsh[128 * LDT];   // >= 32*132 as well

  const int bz = blockIdx.z;
  const int m0 = blockIdx.x << 7;
  const int n0 = blockIdx.y << 7;
  const _Float16* gA = A + (long long)bz * sA + (long long)m0 * lda;
  const _Float16* gB = Bm + (long long)bz * sB;

  const int tid = threadIdx.x;
  const int lane = tid & 31;
  const int wid = tid >> 5;
  const int wm = (wid >> 2) << 6;   // 0,64
  const int wn = (wid & 3) << 5;    // 0,32,64,96
  const int hlf = lane >> 4;        // wave half (0/1)
  const int l16 = lane & 15;

#if USE_TDM
  typedef __attribute__((address_space(3))) _Float16 lds_f16;
  const unsigned ldsA = (unsigned)(unsigned long long)(lds_f16*)&Ash[0];
  const unsigned ldsB = (unsigned)(unsigned long long)(lds_f16*)&Bsh[0];
#endif

  CFrag acc[4][2];
#pragma unroll
  for (int i = 0; i < 4; ++i)
#pragma unroll
    for (int j = 0; j < 2; ++j)
#pragma unroll
      for (int e = 0; e < 8; ++e) acc[i][j].f[e] = 0.f;

  for (int k0 = 0; k0 < Kd; k0 += 32) {
    if (k0 + 32 < Kd) {  // prefetch next A K-slab -> global_prefetch_b8
      __builtin_prefetch(
          (const void*)(gA + (long long)(tid >> 1) * lda + k0 + 32 + ((tid & 1) << 4)), 0, 1);
    }
#if USE_TDM
    // Tensor DMA: one descriptor per tile, issued once (by wave 0 via branch;
    // TDM ignores EXEC). LDS pad codes: 3 -> every 16 dwords, 5 -> every 64
    // dwords; amount code 1 -> 2 dwords (4 halves).
    if (wid == 0) {
      tdm_load_2d_f16(gA + k0, ldsA, 128, 32, lda, 3, 1);
      if (TRANSB)
        tdm_load_2d_f16(gB + (long long)n0 * ldb + k0, ldsB, 128, 32, ldb, 3, 1);
      else
        tdm_load_2d_f16(gB + (long long)k0 * ldb + n0, ldsB, 32, 128, ldb, 5, 1);
      __builtin_amdgcn_s_wait_tensorcnt(0);
    }
#else
    // --- A tile: 128 rows x 32 halves, 8-byte chunks ---
#pragma unroll
    for (int it = 0; it < 4; ++it) {
      int c = tid + (it << 8);
      int row = c >> 3, kc = (c & 7) << 2;
      *(uint2*)&Ash[row * LDT + kc] = *(const uint2*)(gA + (long long)row * lda + k0 + kc);
    }
    if (TRANSB) {
#pragma unroll
      for (int it = 0; it < 4; ++it) {
        int c = tid + (it << 8);
        int nr = c >> 3, kc = (c & 7) << 2;
        *(uint2*)&Bsh[nr * LDT + kc] =
            *(const uint2*)(gB + (long long)(n0 + nr) * ldb + k0 + kc);
      }
    } else {
#pragma unroll
      for (int it = 0; it < 4; ++it) {
        int c = tid + (it << 8);
        int kr = c >> 5, nc = (c & 31) << 2;
        *(uint2*)&Bsh[kr * LDBN + nc] =
            *(const uint2*)(gB + (long long)(k0 + kr) * ldb + n0 + nc);
      }
    }
#endif
    __syncthreads();

    // --- fragments (A and B: lanes0-15:K0-15, 16-31:K16-31)
    HFrag af[4], bf[2];
#pragma unroll
    for (int i = 0; i < 4; ++i) {
      const unsigned short* p = (const unsigned short*)&Ash[(wm + (i << 4) + l16) * LDT];
#pragma unroll
      for (int q = 0; q < 8; ++q) {
        int kb = (q << 1) + (hlf << 4);
        af[i].u[q] = *(const unsigned*)(p + kb);
      }
    }
    if (TRANSB) {
#pragma unroll
      for (int j = 0; j < 2; ++j) {
        const unsigned short* p =
            (const unsigned short*)&Bsh[(wn + (j << 4) + l16) * LDT + (hlf << 4)];
#pragma unroll
        for (int q = 0; q < 8; ++q) bf[j].u[q] = *(const unsigned*)(p + (q << 1));
      }
    } else {
#pragma unroll
      for (int j = 0; j < 2; ++j) {
        const _Float16* p = &Bsh[(hlf << 4) * LDBN + wn + (j << 4) + l16];
#pragma unroll
        for (int q = 0; q < 16; ++q) bf[j].h[q] = p[q * LDBN];
      }
    }
#pragma unroll
    for (int i = 0; i < 4; ++i)
#pragma unroll
      for (int j = 0; j < 2; ++j)
        acc[i][j].v = __builtin_amdgcn_wmma_f32_16x16x32_f16(
            false, af[i].v, false, bf[j].v, (short)0, acc[i][j].v, false, false);
    __syncthreads();
  }

  // --- store: VGPR r holds M = r + 8*half within each 16x16 tile ---
  float* gC = C + (long long)bz * sC + (long long)m0 * N + n0;
#pragma unroll
  for (int i = 0; i < 4; ++i)
#pragma unroll
    for (int j = 0; j < 2; ++j) {
      const int nn = wn + (j << 4) + l16;
#pragma unroll
      for (int r = 0; r < 8; ++r) {
        const int mm = wm + (i << 4) + (hlf << 3) + r;
        gC[(long long)mm * N + nn] = acc[i][j].f[r];
      }
    }
}

// --------------------------- element-wise kernels ---------------------------

// logmap0 scale per row over D (for final co_s/co_c contractions)
__global__ __launch_bounds__(128) void k_row_logscale(const float* __restrict__ rep,
                                                      float* __restrict__ rs) {
  __shared__ float red[4];
  const size_t row = blockIdx.x;
  const int tid = threadIdx.x;
  float v = (tid < DD) ? rep[row * DD + tid] : 0.f;
  float ss = block_sum(v * v, red, tid, 128);
  if (tid == 0) {
    float pn = fmaxf(sqrtf(ss), MINV);
    rs[row] = artanh_(pn) / pn;
  }
}

// proj over the sequence axis of rep^T; optionally also logmap0 -> f16 [B,DP,L]
__global__ __launch_bounds__(256) void k_prep_t(const float* __restrict__ rep,
                                                _Float16* __restrict__ logT,
                                                float* __restrict__ ps, int storeLog) {
  __shared__ float red[8];
  const int bd = blockIdx.x;  // b*128 + d
  const int d = bd & 127;
  const int tid = threadIdx.x;
  float v0 = 0.f, v1 = 0.f;
  if (d < DD) {
    const float* src = rep + ((size_t)(bd >> 7) * SS) * DD + d;
    v0 = src[(size_t)tid * DD];
    v1 = src[(size_t)(tid + 256) * DD];
  }
  float ss = block_sum(v0 * v0 + v1 * v1, red, tid, 256);
  float n = fmaxf(sqrtf(ss), MINV);
  float p = (n > MAXN) ? MAXN / n : 1.f;
  if (tid == 0) ps[bd] = p;
  if (storeLog) {
    float pn = fmaxf(fminf(n, MAXN), MINV);
    float ls = artanh_(pn) / pn;
    logT[(size_t)bd * SS + tid] = (_Float16)(v0 * p * ls);
    logT[(size_t)bd * SS + tid + 256] = (_Float16)(v1 * p * ls);
  }
}

// A1 = logmap0(proj(mobius_matvec(Wl, comment_rep)))  -> f16 [B*T, DP]
__global__ __launch_bounds__(128) void k_mmv_Wl(const float* __restrict__ cr,
                                                const float* __restrict__ Wl,
                                                _Float16* __restrict__ A1) {
  __shared__ float xs[DP];
  __shared__ float red[4];
  const size_t bt = blockIdx.x;
  const int tid = threadIdx.x;
  float xv = (tid < DD) ? cr[bt * DD + tid] : 0.f;
  xs[tid] = xv;
  __syncthreads();
  float x2 = block_sum(xv * xv, red, tid, 128);
  float mx = 0.f;
  if (tid < DD) {
    const float* wrow = Wl + tid * DD;
    for (int d = 0; d < DD; ++d) mx = fmaf(wrow[d], xs[d], mx);
  }
  float m2 = block_sum(mx * mx, red, tid, 128);
  float xn = fmaxf(sqrtf(x2), MINV);
  float mxn = fmaxf(sqrtf(m2), MINV);
  float r = tanhf(mxn / xn * artanh_(xn)) * mx / mxn;  // mx==0 -> r==0
  float r2 = block_sum(r * r, red, tid, 128);
  float n = fmaxf(sqrtf(r2), MINV);
  float p = (n > MAXN) ? MAXN / n : 1.f;
  float pn = fmaxf(fminf(n, MAXN), MINV);
  float ls = artanh_(pn) / pn;
  A1[bt * DP + tid] = (tid < DD) ? (_Float16)(r * p * ls) : (_Float16)0.f;
}

// G^T = mobius_matvec(W[K,D], proj'd rep rows), scattered to [B,K,L] f32
__global__ __launch_bounds__(128) void k_mmv_W(const float* __restrict__ rep,
                                               const float* __restrict__ ps,
                                               const float* __restrict__ W,
                                               float* __restrict__ Gt) {
  __shared__ float xs[DP];
  __shared__ float red[4];
  const int bs = blockIdx.x;        // b*512 + s
  const int b = bs >> 9, s = bs & 511;
  const int tid = threadIdx.x;
  float xv = (tid < DD) ? rep[(size_t)bs * DD + tid] * ps[(b << 7) + tid] : 0.f;
  xs[tid] = xv;
  __syncthreads();
  float x2 = block_sum(xv * xv, red, tid, 128);
  const float* wrow = W + tid * DD;  // K=128 rows, one per thread
  float mx = 0.f;
  for (int d = 0; d < DD; ++d) mx = fmaf(wrow[d], xs[d], mx);
  float m2 = block_sum(mx * mx, red, tid, 128);
  float xn = fmaxf(sqrtf(x2), MINV);
  float mxn = fmaxf(sqrtf(m2), MINV);
  float r = tanhf(mxn / xn * artanh_(xn)) * mx / mxn;
  Gt[((size_t)b * KK + tid) * SS + s] = r;
}

// L_e = logmap0(proj(expmap0(tanh(raw))))  per row of 512, f32 -> f16
__global__ __launch_bounds__(256) void k_rowLe(const float* __restrict__ tmp,
                                               _Float16* __restrict__ Le) {
  __shared__ float red[8];
  const size_t row = blockIdx.x;
  const int tid = threadIdx.x;
  const float* x = tmp + row * SS;
  float v0 = tanhf(x[tid]), v1 = tanhf(x[tid + 256]);
  float ss = block_sum(v0 * v0 + v1 * v1, red, tid, 256);
  float n1 = fmaxf(sqrtf(ss), MINV);
  float s1 = tanhf(n1) / n1;                      // expmap0
  float n2 = fmaxf(tanhf(n1), MINV);
  float s2 = (n2 > MAXN) ? MAXN / n2 : 1.f;       // proj
  float n3 = fmaxf(fminf(n2, MAXN), MINV);
  float s3 = artanh_(n3) / n3;                    // logmap0
  float sc = s1 * s2 * s3;
  Le[row * SS + tid] = (_Float16)(v0 * sc);
  Le[row * SS + tid + 256] = (_Float16)(v1 * sc);
}

// in-place proj over row; also emit f16 logmap0(row)
__global__ __launch_bounds__(256) void k_proj_log_rows(float* __restrict__ G,
                                                       _Float16* __restrict__ lG) {
  __shared__ float red[8];
  const size_t row = blockIdx.x;
  const int tid = threadIdx.x;
  float* x = G + row * SS;
  float v0 = x[tid], v1 = x[tid + 256];
  float ss = block_sum(v0 * v0 + v1 * v1, red, tid, 256);
  float n = fmaxf(sqrtf(ss), MINV);
  float p = (n > MAXN) ? MAXN / n : 1.f;
  float pn = fmaxf(fminf(n, MAXN), MINV);
  float ls = artanh_(pn) / pn;
  float g0 = v0 * p, g1 = v1 * p;
  x[tid] = g0;
  x[tid + 256] = g1;
  lG[row * SS + tid] = (_Float16)(g0 * ls);
  lG[row * SS + tid + 256] = (_Float16)(g1 * ls);
}

// in-place proj(expmap0(row))
__global__ __launch_bounds__(256) void k_exp_proj_rows(float* __restrict__ X) {
  __shared__ float red[8];
  const size_t row = blockIdx.x;
  const int tid = threadIdx.x;
  float* x = X + row * SS;
  float v0 = x[tid], v1 = x[tid + 256];
  float ss = block_sum(v0 * v0 + v1 * v1, red, tid, 256);
  float n1 = fmaxf(sqrtf(ss), MINV);
  float a = tanhf(n1) / n1;
  float n2 = fmaxf(tanhf(n1), MINV);
  float s2 = (n2 > MAXN) ? MAXN / n2 : 1.f;
  float sc = a * s2;
  x[tid] = v0 * sc;
  x[tid + 256] = v1 * sc;
}

// Y := proj(expmap0(tanh(logmap0(proj(mobius_add(X, Y))))))   (c = 1)
__global__ __launch_bounds__(256) void k_hyp_combine(const float* __restrict__ Xb,
                                                     float* __restrict__ Yb) {
  __shared__ float red[8];
  const size_t row = blockIdx.x;
  const int tid = threadIdx.x;
  const float* xp = Xb + row * SS;
  float* yp = Yb + row * SS;
  float x0 = xp[tid], x1 = xp[tid + 256];
  float y0 = yp[tid], y1 = yp[tid + 256];
  float x2 = block_sum(x0 * x0 + x1 * x1, red, tid, 256);
  float y2 = block_sum(y0 * y0 + y1 * y1, red, tid, 256);
  float xy = block_sum(x0 * y0 + x1 * y1, red, tid, 256);
  float ca = 1.f + 2.f * xy + y2;
  float cb = 1.f - x2;
  float den = fmaxf(1.f + 2.f * xy + x2 * y2, MINV);
  float h0 = (ca * x0 + cb * y0) / den;
  float h1 = (ca * x1 + cb * y1) / den;
  float hs = block_sum(h0 * h0 + h1 * h1, red, tid, 256);
  float n = fmaxf(sqrtf(hs), MINV);
  float p = (n > MAXN) ? MAXN / n : 1.f;
  float pn = fmaxf(fminf(n, MAXN), MINV);
  float ls = artanh_(pn) / pn;
  float t0 = tanhf(h0 * p * ls), t1 = tanhf(h1 * p * ls);
  float ts = block_sum(t0 * t0 + t1 * t1, red, tid, 256);
  float nt = fmaxf(sqrtf(ts), MINV);
  float a = tanhf(nt) / nt;
  float n2 = fmaxf(tanhf(nt), MINV);
  float s2 = (n2 > MAXN) ? MAXN / n2 : 1.f;
  yp[tid] = t0 * a * s2;
  yp[tid + 256] = t1 * a * s2;
}

// per (b,s): mobius_matvec with 1xK weight down columns of H [B,K,512]
__global__ __launch_bounds__(256) void k_att_pre(const float* __restrict__ H,
                                                 const float* __restrict__ wv,
                                                 float* __restrict__ outp) {
  const int idx = blockIdx.x * 256 + threadIdx.x;  // b*512 + s
  const int b = idx >> 9, s = idx & 511;
  const float* base = H + ((size_t)b * KK) * SS + s;
  float mx = 0.f, x2 = 0.f;
#pragma unroll 4
  for (int k = 0; k < KK; ++k) {
    float h = base[(size_t)k * SS];
    mx = fmaf(h, wv[k], mx);
    x2 = fmaf(h, h, x2);
  }
  float xn = fmaxf(sqrtf(x2), MINV);
  float mxn = fmaxf(fabsf(mx), MINV);
  outp[idx] = tanhf(mxn / xn * artanh_(xn)) * mx / mxn;
}

// proj -> logmap0 -> softmax -> expmap0 -> proj, plus logmap0 of the result
__global__ __launch_bounds__(256) void k_att_final(const float* __restrict__ pre,
                                                   float* __restrict__ Aout,
                                                   float* __restrict__ logA) {
  __shared__ float red[8];
  const size_t b = blockIdx.x;
  const int tid = threadIdx.x;
  const float* x = pre + b * SS;
  float v0 = x[tid], v1 = x[tid + 256];
  float ss = block_sum(v0 * v0 + v1 * v1, red, tid, 256);
  float n = fmaxf(sqrtf(ss), MINV);
  float p = (n > MAXN) ? MAXN / n : 1.f;
  float pn = fmaxf(fminf(n, MAXN), MINV);
  float ls = artanh_(pn) / pn;
  float l0 = v0 * p * ls, l1 = v1 * p * ls;
  float mx = block_max(fmaxf(l0, l1), red, tid, 256);
  float e0 = expf(l0 - mx), e1 = expf(l1 - mx);
  float es = block_sum(e0 + e1, red, tid, 256);
  float sm0 = e0 / es, sm1 = e1 / es;
  float ms = block_sum(sm0 * sm0 + sm1 * sm1, red, tid, 256);
  float ns = fmaxf(sqrtf(ms), MINV);
  float a = tanhf(ns) / ns;
  float n2 = fmaxf(tanhf(ns), MINV);
  float s2 = (n2 > MAXN) ? MAXN / n2 : 1.f;
  float A0 = sm0 * a * s2, A1v = sm1 * a * s2;
  Aout[b * SS + tid] = A0;
  Aout[b * SS + tid + 256] = A1v;
  float n3 = fmaxf(fminf(n2, MAXN), MINV);
  float ls2 = artanh_(n3) / n3;
  logA[b * SS + tid] = A0 * ls2;
  logA[b * SS + tid + 256] = A1v * ls2;
}

// co = logmap0(A) @ logmap0(rep) : one block per batch, thread per feature
__global__ __launch_bounds__(128) void k_final_dot(const float* __restrict__ rep,
                                                   const float* __restrict__ rs,
                                                   const float* __restrict__ logA,
                                                   float* __restrict__ outp, int colOff) {
  const int b = blockIdx.x, d = threadIdx.x;
  if (d >= DD) return;
  const float* repb = rep + (size_t)b * SS * DD;
  const float* la = logA + (size_t)b * SS;
  const float* rsb = rs + (size_t)b * SS;
  float acc = 0.f;
  for (int s = 0; s < SS; ++s) acc = fmaf(la[s] * rsb[s], repb[(size_t)s * DD + d], acc);
  outp[b * (2 * DD) + colOff + d] = acc;
}

// --------------------------------- launch -----------------------------------

extern "C" void kernel_launch(void* const* d_in, const int* in_sizes, int n_in,
                              void* d_out, int out_size, void* d_ws, size_t ws_size,
                              hipStream_t stream) {
  (void)in_sizes; (void)n_in; (void)out_size; (void)ws_size;
  const float* sr  = (const float*)d_in[0];  // [B,S,D]
  const float* cr  = (const float*)d_in[1];  // [B,T,D]
  const float* Wl  = (const float*)d_in[2];  // [D,D]
  const float* Wc  = (const float*)d_in[3];  // [K,D]
  const float* Ws  = (const float*)d_in[4];  // [K,D]
  const float* whs = (const float*)d_in[5];  // [1,K]
  const float* whc = (const float*)d_in[6];  // [1,K]
  float* out = (float*)d_out;                // [B*2D | B*S | B*T]

  // ---- workspace layout (~304 MB) ----
  char* w = (char*)d_ws;
  constexpr size_t SZ_LOGST = (size_t)BB * DP * SS * 2;    // f16 [B,DP,S]
  constexpr size_t SZ_A1    = (size_t)BB * SS * DP * 2;    // f16 [B,T,DP]
  constexpr size_t SZ_LE    = (size_t)BB * SS * SS * 2;    // f16 [B,T,S]
  constexpr size_t SZ_BIG   = (size_t)BB * SS * SS * 4;    // f32 [B,T,S] (raw L)
  constexpr size_t O_LOGST = 0;
  constexpr size_t O_A1   = O_LOGST + SZ_LOGST;
  constexpr size_t O_LE   = O_A1 + SZ_A1;
  constexpr size_t O_BIG  = O_LE + SZ_LE;
  // after L_e is extracted, the BIG region is reused:
  constexpr size_t O_GA   = O_BIG;                               // f32 [B,K,S]
  constexpr size_t O_GB   = O_GA + (size_t)BB * KK * SS * 4;     // f32 [B,K,T]
  constexpr size_t O_LGA  = O_GB + (size_t)BB * KK * SS * 4;     // f16 [B,K,S]
  constexpr size_t O_LGB  = O_LGA + (size_t)BB * KK * SS * 2;    // f16 [B,K,T]
  constexpr size_t O_TMP2 = O_BIG + SZ_BIG;                      // f32 [B,K,S]
  constexpr size_t O_TMP3 = O_TMP2 + (size_t)BB * KK * SS * 4;   // f32 [B,K,T]
  constexpr size_t O_PS_S = O_TMP3 + (size_t)BB * KK * SS * 4;
  constexpr size_t O_PS_C = O_PS_S + (size_t)BB * DP * 4;
  constexpr size_t O_RS_S = O_PS_C + (size_t)BB * DP * 4;
  constexpr size_t O_RS_C = O_RS_S + (size_t)BB * SS * 4;
  constexpr size_t O_ASP  = O_RS_C + (size_t)BB * SS * 4;
  constexpr size_t O_ACP  = O_ASP + (size_t)BB * SS * 4;
  constexpr size_t O_LAS  = O_ACP + (size_t)BB * SS * 4;
  constexpr size_t O_LAC  = O_LAS + (size_t)BB * SS * 4;

  _Float16* log_s_t = (_Float16*)(w + O_LOGST);
  _Float16* A1      = (_Float16*)(w + O_A1);
  _Float16* Le      = (_Float16*)(w + O_LE);
  float*    tmpL    = (float*)(w + O_BIG);
  float*    Ga      = (float*)(w + O_GA);
  float*    Gb      = (float*)(w + O_GB);
  _Float16* lGa     = (_Float16*)(w + O_LGA);
  _Float16* lGb     = (_Float16*)(w + O_LGB);
  float*    tmp2    = (float*)(w + O_TMP2);
  float*    tmp3    = (float*)(w + O_TMP3);
  float*    ps_s    = (float*)(w + O_PS_S);
  float*    ps_c    = (float*)(w + O_PS_C);
  float*    rs_sr   = (float*)(w + O_RS_S);
  float*    rs_cr   = (float*)(w + O_RS_C);
  float*    As_pre  = (float*)(w + O_ASP);
  float*    Ac_pre  = (float*)(w + O_ACP);
  float*    logAs   = (float*)(w + O_LAS);
  float*    logAc   = (float*)(w + O_LAC);

  // ---- stage A: radial prep ----
  k_row_logscale<<<BB * SS, 128, 0, stream>>>(sr, rs_sr);
  k_row_logscale<<<BB * SS, 128, 0, stream>>>(cr, rs_cr);
  k_prep_t<<<BB * DP, 256, 0, stream>>>(sr, log_s_t, ps_s, 1);   // logmap0(proj(sr^T)) + ps_s
  k_prep_t<<<BB * DP, 256, 0, stream>>>(cr, log_s_t, ps_c, 0);   // ps_c only
  k_mmv_Wl<<<BB * SS, 128, 0, stream>>>(cr, Wl, A1);

  // ---- stage B: affinity L_e (WMMA GEMM1, Kd = DP = 128) ----
  k_gemm_f16<0><<<dim3(4, 4, BB), 256, 0, stream>>>(
      A1, (long long)SS * DP, DP, log_s_t, (long long)DP * SS, SS,
      tmpL, (long long)SS * SS, SS, DP);
  k_rowLe<<<BB * SS, 256, 0, stream>>>(tmpL, Le);

  // ---- stage C: Ga (=Hs_a=Hc_b-pre), Gb (=Hc_a=Hs_b-pre) ----
  k_mmv_W<<<BB * SS, 128, 0, stream>>>(sr, ps_s, Ws, Ga);
  k_mmv_W<<<BB * SS, 128, 0, stream>>>(cr, ps_c, Wc, Gb);
  k_proj_log_rows<<<BB * KK, 256, 0, stream>>>(Ga, lGa);
  k_proj_log_rows<<<BB * KK, 256, 0, stream>>>(Gb, lGb);

  // ---- stage D: cross terms (WMMA GEMM2/3, Kd = 512) ----
  k_gemm_f16<0><<<dim3(1, 4, BB), 256, 0, stream>>>(
      lGb, (long long)KK * SS, SS, Le, (long long)SS * SS, SS,
      tmp2, (long long)KK * SS, SS, SS);
  k_exp_proj_rows<<<BB * KK, 256, 0, stream>>>(tmp2);            // -> Hs_b
  k_gemm_f16<1><<<dim3(1, 4, BB), 256, 0, stream>>>(
      lGa, (long long)KK * SS, SS, Le, (long long)SS * SS, SS,
      tmp3, (long long)KK * SS, SS, SS);
  k_exp_proj_rows<<<BB * KK, 256, 0, stream>>>(tmp3);            // -> Hc_b

  // ---- stage E: mobius add + activation, attention weights ----
  k_hyp_combine<<<BB * KK, 256, 0, stream>>>(Ga, tmp2);          // tmp2 := Hs
  k_hyp_combine<<<BB * KK, 256, 0, stream>>>(Gb, tmp3);          // tmp3 := Hc
  k_att_pre<<<(BB * SS) / 256, 256, 0, stream>>>(tmp2, whs, As_pre);
  k_att_pre<<<(BB * SS) / 256, 256, 0, stream>>>(tmp3, whc, Ac_pre);
  k_att_final<<<BB, 256, 0, stream>>>(As_pre, out + BB * 2 * DD, logAs);
  k_att_final<<<BB, 256, 0, stream>>>(Ac_pre, out + BB * 2 * DD + BB * SS, logAc);

  // ---- stage F: co_s / co_c ----
  k_final_dot<<<BB, 128, 0, stream>>>(sr, rs_sr, logAs, out, 0);
  k_final_dot<<<BB, 128, 0, stream>>>(cr, rs_cr, logAc, out, DD);
}
